// HNNLayer_59124519796864
// MI455X (gfx1250) — matrix operand order repaired and checked by reading, not compiled, not run on hardware
//
#include <hip/hip_runtime.h>

// CDNA5 WMMA vector types (probe-verified signatures)
typedef __attribute__((ext_vector_type(16))) __bf16 v16bf;
typedef __attribute__((ext_vector_type(8)))  float  v8f;

namespace {
constexpr int   TM     = 32;          // rows per workgroup
constexpr int   TSTb   = 264;         // t row stride in bf16 elems (528B, 16B aligned, bank-spread)
constexpr int   XST    = 260;         // xt row stride in floats (aliases thi/tlo region)
constexpr int   WSTb   = 40;          // w slab row stride in bf16 elems (80B)
constexpr int   WSLAB  = 256 * WSTb;  // one hi or lo slab, bf16 elems
constexpr float EPSf   = 1e-15f;
constexpr float BOUNDf = 1.0f - 1e-5f;
constexpr float LNEPS  = 1e-5f;
}

// persistent split weight (bf16 hi + residual), produced once per launch by w_split_kernel;
// 2 x 128KB -> permanently resident in the 192MB L2
__device__ __bf16 g_whi[256 * 256];
__device__ __bf16 g_wlo[256 * 256];

// wave32 butterfly all-reduce (every lane gets the full sum)
__device__ __forceinline__ float wred_sum(float v) {
#pragma unroll
  for (int off = 16; off > 0; off >>= 1) v += __shfl_xor(v, off, 32);
  return v;
}

// split fp32 into bf16 hi + bf16 residual (bf16x3 fp32-GEMM emulation)
__device__ __forceinline__ void bf_split(float x, __bf16& hi, __bf16& lo) {
  hi = (__bf16)x;
  lo = (__bf16)(x - (float)hi);
}

union Frag16 { v16bf v; uint4 q[2]; };   // one WMMA operand = 2 x 16B LDS loads
union BF4   { __bf16 b[4]; uint2 u; };   // 4 packed bf16 = one 8B store

// ---------------- prep: W (fp32) -> g_whi/g_wlo (split bf16), 64 blocks ----------------
__global__ __launch_bounds__(256) void w_split_kernel(const float* __restrict__ w) {
  const int i = blockIdx.x * 256 + threadIdx.x;     // 16384 threads x float4
  const float4 v = ((const float4*)w)[i];
  BF4 hi, lo;
  bf_split(v.x, hi.b[0], lo.b[0]);
  bf_split(v.y, hi.b[1], lo.b[1]);
  bf_split(v.z, hi.b[2], lo.b[2]);
  bf_split(v.w, hi.b[3], lo.b[3]);
  *(uint2*)(g_whi + i * 4) = hi.u;
  *(uint2*)(g_wlo + i * 4) = lo.u;
}

// async-DMA one 256x32 split-bf16 W slab into LDS (per-lane 16B ops, ASYNCcnt-tracked)
__device__ __forceinline__ void issue_slab_copy(int ks, __bf16* wbuf, int bufsel, int tid) {
#pragma unroll
  for (int p = 0; p < 4; ++p) {
    const int o = (tid >> 2) + p * 64;              // weight row 0..255
    const int j = (tid & 3) * 8;                    // 8 bf16 = 16B chunk
    const unsigned lhi =
        (unsigned)(uintptr_t)(wbuf + bufsel + o * WSTb + j);
    const unsigned llo =
        (unsigned)(uintptr_t)(wbuf + bufsel + WSLAB + o * WSTb + j);
    const unsigned long long ghi =
        (unsigned long long)(uintptr_t)(g_whi + o * 256 + ks * 32 + j);
    const unsigned long long glo =
        (unsigned long long)(uintptr_t)(g_wlo + o * 256 + ks * 32 + j);
    asm volatile("global_load_async_to_lds_b128 %0, %1, off" :: "v"(lhi), "v"(ghi) : "memory");
    asm volatile("global_load_async_to_lds_b128 %0, %1, off" :: "v"(llo), "v"(glo) : "memory");
  }
}

__global__ __launch_bounds__(256) void hnn_layer_fused(
    const float* __restrict__ h, const float* __restrict__ bias,
    const float* __restrict__ gamma, const float* __restrict__ beta,
    float* __restrict__ out) {
  extern __shared__ char smem[];
  // [TM][TSTb] bf16 hi/lo of t; region reused as fp32 xt[TM][XST] after the GEMM
  __bf16* thi = (__bf16*)smem;
  __bf16* tlo = thi + TM * TSTb;
  float*  xt  = (float*)smem;
  // double-buffered split W k-slabs: [buf0: hi|lo][buf1: hi|lo], each [256][WSTb]
  __bf16* wbuf = tlo + TM * TSTb;

  const int tid  = threadIdx.x;
  const int lane = tid & 31;
  const int wv   = tid >> 5;   // wave id 0..7
  const int half = lane >> 4;  // 0/1 (lane group per WMMA layout tables)
  const int m16  = lane & 15;
  const long row0 = (long)blockIdx.x * TM;

  // kick off slab 0 DMA immediately; it overlaps the transcendental-heavy Phase A
  issue_slab_copy(0, wbuf, 0, tid);

  // per-lane column set e = lane + 32*i (coalesced), invariant across rows
  float gammav[8], betav[8], biasv[8];
#pragma unroll
  for (int i = 0; i < 8; ++i) {
    const int e = lane + 32 * i;
    gammav[i] = gamma[e];
    betav[i]  = beta[e];
    biasv[i]  = bias[e];
  }

  // ============ Phase A: t = logmap0(expmap0(LN(logmap0(h)))) -> split bf16 LDS ============
#pragma unroll
  for (int q = 0; q < 4; ++q) {
    const int rl = wv * 4 + q;                       // 8 waves x 4 rows = 32 rows
    const float* __restrict__ hr = h + (row0 + rl) * 256;
    float x[8];
    float s = 0.f;
#pragma unroll
    for (int i = 0; i < 8; ++i) { x[i] = hr[lane + 32 * i]; s += x[i] * x[i]; }
    float n  = fmaxf(sqrtf(wred_sum(s)), EPSf);
    float k0 = atanhf(fminf(n, BOUNDf)) / n;         // logmap0 scale
    s = 0.f;
#pragma unroll
    for (int i = 0; i < 8; ++i) { x[i] *= k0; s += x[i]; }
    const float mu = wred_sum(s) * (1.0f / 256.0f);
    s = 0.f;
#pragma unroll
    for (int i = 0; i < 8; ++i) { x[i] -= mu; s += x[i] * x[i]; }
    const float inv = rsqrtf(wred_sum(s) * (1.0f / 256.0f) + LNEPS);
    s = 0.f;
#pragma unroll
    for (int i = 0; i < 8; ++i) {
      x[i] = x[i] * inv * gammav[i] + betav[i];
      s += x[i] * x[i];
    }
    float ny = fmaxf(sqrtf(wred_sum(s)), EPSf);
    float k1 = tanhf(ny) / ny;                       // expmap0 scale
    s = 0.f;
#pragma unroll
    for (int i = 0; i < 8; ++i) { x[i] *= k1; s += x[i] * x[i]; }
    float nh = fmaxf(sqrtf(wred_sum(s)), EPSf);
    float k2 = atanhf(fminf(nh, BOUNDf)) / nh;       // logmap0 scale
#pragma unroll
    for (int i = 0; i < 8; ++i) {
      __bf16 hi, lo;
      bf_split(x[i] * k2, hi, lo);                   // split ONCE per element
      thi[rl * TSTb + lane + 32 * i] = hi;
      tlo[rl * TSTb + lane + 32 * i] = lo;
    }
  }

  // ============ Phase B: xt = t @ W^T, split-bf16 WMMA, double-buffered async slabs ============
  const int rwb = (wv >> 2) * 16;  // wave row-tile base (0 or 16)
  const int ob0 = (wv & 3) * 64;   // wave col base (4 tiles of 16)

  const v8f vzero = {0.f, 0.f, 0.f, 0.f, 0.f, 0.f, 0.f, 0.f};
  v8f acc[4];
#pragma unroll
  for (int t = 0; t < 4; ++t) acc[t] = vzero;

  for (int ks = 0; ks < 8; ++ks) {
    asm volatile("s_wait_asynccnt 0x0" ::: "memory");  // our slab-ks DMA landed
    __syncthreads();                                   // everyone's landed; other buffer free
    if (ks < 7) {
      issue_slab_copy(ks + 1, wbuf, ((ks + 1) & 1) * 2 * WSLAB, tid);  // hide behind WMMA
      if (ks < 6)  // deep prefetch: pull slab ks+2 toward L2 (global_prefetch_b8)
        __builtin_prefetch(g_whi + ((tid >> 2) + (tid & 3) * 64) * 256 + (ks + 2) * 32, 0, 1);
    }
    // offset-based buffer select on the smem-rooted pointer (keeps AS(3) -> ds_load_b128)
    const int bsel = (ks & 1) * 2 * WSLAB;

    // A fragments: two 16B loads each (K runs [kb,kb+8) and [kb+16,kb+24))
    const __bf16* arowh = thi + (rwb + m16) * TSTb;
    const __bf16* arowl = tlo + (rwb + m16) * TSTb;
    const int kb = ks * 32 + half * 8;
    Frag16 ah, al;
    ah.q[0] = *(const uint4*)(arowh + kb);
    ah.q[1] = *(const uint4*)(arowh + kb + 16);
    al.q[0] = *(const uint4*)(arowl + kb);
    al.q[1] = *(const uint4*)(arowl + kb + 16);

#pragma unroll
    for (int tn = 0; tn < 4; ++tn) {
      // B fragment: lane m16 = output column o; K runs [half*8,+8) and [16+half*8,+8)
      const int ob = bsel + (ob0 + tn * 16 + m16) * WSTb + half * 8;
      Frag16 bh, bl;
      bh.q[0] = *(const uint4*)(wbuf + ob);
      bh.q[1] = *(const uint4*)(wbuf + ob + 16);
      bl.q[0] = *(const uint4*)(wbuf + ob + WSLAB);
      bl.q[1] = *(const uint4*)(wbuf + ob + WSLAB + 16);
      // fp32-accurate GEMM: hi*hi + hi*lo + lo*hi (drop lo*lo)
      acc[tn] = __builtin_amdgcn_wmma_f32_16x16x32_bf16(false, ah.v, false, bh.v, (short)0, acc[tn], false, false);
      acc[tn] = __builtin_amdgcn_wmma_f32_16x16x32_bf16(false, ah.v, false, bl.v, (short)0, acc[tn], false, false);
      acc[tn] = __builtin_amdgcn_wmma_f32_16x16x32_bf16(false, al.v, false, bh.v, (short)0, acc[tn], false, false);
    }
  }

  __syncthreads();  // all thi/tlo reads complete -> region reused as fp32 xt
  // C/D f32 16x16 layout: M = vgpr + 8*(lane>=16), N = lane%16
#pragma unroll
  for (int tn = 0; tn < 4; ++tn)
#pragma unroll
    for (int v = 0; v < 8; ++v)
      xt[(rwb + v + 8 * half) * XST + ob0 + tn * 16 + m16] = acc[tn][v];
  __syncthreads();

  // ============ Phase C: expmap0 -> transp0 -> expmap -> act chain ============
#pragma unroll
  for (int q = 0; q < 4; ++q) {
    const int rl = wv * 4 + q;
    float xv[8];
    float s = 0.f;
#pragma unroll
    for (int i = 0; i < 8; ++i) { xv[i] = xt[rl * XST + lane + 32 * i]; s += xv[i] * xv[i]; }
    float nx = fmaxf(sqrtf(wred_sum(s)), EPSf);
    float kx = tanhf(nx) / nx;                       // res = expmap0(xt)
    float res[8], bb[8];
    float x2 = 0.f;
#pragma unroll
    for (int i = 0; i < 8; ++i) { res[i] = xv[i] * kx; x2 += res[i] * res[i]; }
    x2 = wred_sum(x2);
    const float sfac = 1.0f - x2;                    // transp0: b = bias*(1-||res||^2)
    float y2 = 0.f;
#pragma unroll
    for (int i = 0; i < 8; ++i) { bb[i] = biasv[i] * sfac; y2 += bb[i] * bb[i]; }
    y2 = wred_sum(y2);
    const float nb  = fmaxf(sqrtf(y2), EPSf);
    const float lam = 2.0f / fmaxf(1.0f - x2, EPSf);
    const float kse = tanhf(0.5f * lam * nb) / nb;   // expmap second term scale
    float xy = 0.f, s2 = 0.f;
#pragma unroll
    for (int i = 0; i < 8; ++i) { bb[i] *= kse; xy += res[i] * bb[i]; s2 += bb[i] * bb[i]; }
    xy = wred_sum(xy);
    s2 = wred_sum(s2);
    // mobius_add(res, second)
    const float ca  = 1.0f + 2.0f * xy + s2;
    const float cb  = 1.0f - x2;
    const float den = 1.0f / fmaxf(1.0f + 2.0f * xy + x2 * s2, EPSf);
    float mv[8];
    float m2 = 0.f;
#pragma unroll
    for (int i = 0; i < 8; ++i) { mv[i] = (ca * res[i] + cb * bb[i]) * den; m2 += mv[i] * mv[i]; }
    float nm = fmaxf(sqrtf(wred_sum(m2)), EPSf);
    float km = atanhf(fminf(nm, BOUNDf)) / nm;       // logmap0
    float u2 = 0.f;
#pragma unroll
    for (int i = 0; i < 8; ++i) { mv[i] = fmaxf(mv[i] * km, 0.f); u2 += mv[i] * mv[i]; }  // relu
    float nu = fmaxf(sqrtf(wred_sum(u2)), EPSf);
    float ku = tanhf(nu) / nu;                       // expmap0 (c_out = 1)
    float* __restrict__ orow = out + (row0 + rl) * 256;
#pragma unroll
    for (int i = 0; i < 8; ++i) orow[lane + 32 * i] = mv[i] * ku;
  }
}

extern "C" void kernel_launch(void* const* d_in, const int* in_sizes, int n_in,
                              void* d_out, int out_size, void* d_ws, size_t ws_size,
                              hipStream_t stream) {
  (void)n_in; (void)out_size; (void)d_ws; (void)ws_size;
  const float* h      = (const float*)d_in[0];
  const float* weight = (const float*)d_in[1];
  const float* bias   = (const float*)d_in[2];
  const float* gamma  = (const float*)d_in[3];
  const float* beta   = (const float*)d_in[4];
  float* out = (float*)d_out;

  const int N      = in_sizes[0] / 256;       // rows
  const int blocks = N / TM;                  // 8192 for N=262144

  // thi/tlo (2*TM*TSTb bf16) + 2 double-buffered hi/lo slabs (4*WSLAB bf16)
  const size_t shmem = (size_t)(2 * TM * TSTb * 2 + 4 * WSLAB * 2);

  // 1) split W into persistent bf16 hi/lo (L2-resident), 2) fused layer
  hipLaunchKernelGGL(w_split_kernel, dim3(64), dim3(256), 0, stream, weight);
  (void)hipFuncSetAttribute((const void*)hnn_layer_fused,
                            hipFuncAttributeMaxDynamicSharedMemorySize, (int)shmem);
  hipLaunchKernelGGL(hnn_layer_fused, dim3(blocks), dim3(256), shmem, stream,
                     h, bias, gamma, beta, out);
}
